// L2PANConv_84859963834444
// MI455X (gfx1250) — matrix operand order repaired and checked by reading, not compiled, not run on hardware
//
#include <hip/hip_runtime.h>
#include <math.h>

typedef __attribute__((ext_vector_type(2)))  float    v2f;
typedef __attribute__((ext_vector_type(8)))  float    v8f;
typedef __attribute__((ext_vector_type(16))) _Float16 v16h;

#define NNODES 2048
#define INC    128
#define OUTC   64
#define HIDC   6400
#define LPOW   5

// ---------------------------------------------------------------------------
// Exact fp32 WMMA GEMM (V_WMMA_F32_16X16X4_F32):  C = A @ B  (or A @ B^T)
// Used for the value-carrying GEMMs (arbitrary float operands).
// Block tile 128x64, 8 waves in 4x2, wave tile 32x32. M%128==N%64==K%32==0.
// ---------------------------------------------------------------------------
template <bool TRANSB>
__global__ __launch_bounds__(256) void gemm_f32_wmma(
    const float* __restrict__ A, const float* __restrict__ B,
    float* __restrict__ C, int M, int Nn, int K) {
  constexpr int BM = 128, BN = 64, BK = 32;
  __shared__ float As[BM][BK + 1];
  __shared__ float Bs[BK][BN + 2];

  const int tid  = threadIdx.x;
  const int lane = tid & 31;
  const int wave = tid >> 5;
  const int half = lane >> 4;
  const int l16  = lane & 15;
  const int wm   = wave >> 1;
  const int wn   = wave & 1;
  const int blockM = blockIdx.y * BM;
  const int blockN = blockIdx.x * BN;

  v8f acc[2][2] = {};

  for (int k0 = 0; k0 < K; k0 += BK) {
#pragma unroll
    for (int i = 0; i < (BM * BK) / 256; ++i) {
      int idx = tid + i * 256;
      int r = idx >> 5, c = idx & 31;
      As[r][c] = A[(size_t)(blockM + r) * K + (k0 + c)];
    }
#pragma unroll
    for (int i = 0; i < (BK * BN) / 256; ++i) {
      int idx = tid + i * 256;
      int r = idx >> 6, c = idx & 63;
      Bs[r][c] = TRANSB ? B[(size_t)(blockN + c) * K + (k0 + r)]
                        : B[(size_t)(k0 + r) * Nn + (blockN + c)];
    }
    if (k0 + BK < K)
      __builtin_prefetch(&A[(size_t)(blockM + (tid >> 1)) * K + (k0 + BK)], 0, 1);
    __syncthreads();

#pragma unroll
    for (int kk = 0; kk < BK; kk += 4) {
      v2f afr[2], bfr[2];
#pragma unroll
      for (int tm = 0; tm < 2; ++tm) {
        int row = wm * 32 + tm * 16 + l16;
        afr[tm][0] = As[row][kk + 2 * half + 0];
        afr[tm][1] = As[row][kk + 2 * half + 1];
      }
#pragma unroll
      for (int tn = 0; tn < 2; ++tn) {
        int col = wn * 32 + tn * 16 + l16;
        bfr[tn][0] = Bs[kk + 2 * half + 0][col];
        bfr[tn][1] = Bs[kk + 2 * half + 1][col];
      }
#pragma unroll
      for (int tm = 0; tm < 2; ++tm)
#pragma unroll
        for (int tn = 0; tn < 2; ++tn)
          acc[tm][tn] = __builtin_amdgcn_wmma_f32_16x16x4_f32(
              false, afr[tm], false, bfr[tn], (short)0, acc[tm][tn],
              false, false);
    }
    __syncthreads();
  }

#pragma unroll
  for (int tm = 0; tm < 2; ++tm)
#pragma unroll
    for (int tn = 0; tn < 2; ++tn) {
      int col = blockN + wn * 32 + tn * 16 + l16;
#pragma unroll
      for (int g = 0; g < 8; ++g) {
        int row = blockM + wm * 32 + tm * 16 + g + 8 * half;
        C[(size_t)row * Nn + col] = acc[tm][tn][g];
      }
    }
}

// ---------------------------------------------------------------------------
// Split-f16 power GEMM (V_WMMA_F32_16X16X32_F16):
//   SPLIT:  C = (2048*Ahi + Alo) @ B   with Ahi/Alo/B f16, f32 accumulate.
//   !SPLIT: C = Alo @ B               (binary x binary for A^2).
// Exact for non-negative integer A < 2^22 against binary B.
// Block tile 128x64, wave tile 32x32, BK=32 (one f16 WMMA k-step per stage).
// ---------------------------------------------------------------------------
template <bool SPLIT>
__global__ __launch_bounds__(256) void gemm_pow_f16_wmma(
    const _Float16* __restrict__ Ahi, const _Float16* __restrict__ Alo,
    const _Float16* __restrict__ B, float* __restrict__ C,
    int M, int Nn, int K) {
  constexpr int BM = 128, BN = 64, BK = 32;
  __shared__ _Float16 Ah[BM][BK + 2];   // stride 34 halves = 17 banks (odd) -> clean
  __shared__ _Float16 Al[BM][BK + 2];
  __shared__ _Float16 Bsh[BK][BN + 2];

  const int tid  = threadIdx.x;
  const int lane = tid & 31;
  const int wave = tid >> 5;
  const int half = lane >> 4;
  const int l16  = lane & 15;
  const int wm   = wave >> 1;
  const int wn   = wave & 1;
  const int blockM = blockIdx.y * BM;
  const int blockN = blockIdx.x * BN;

  v8f accL[2][2] = {};
  v8f accH[2][2] = {};

  for (int k0 = 0; k0 < K; k0 += BK) {
    // ---- stage A (hi+lo) and B tiles: 4096 halves each A mat, 2048 B ----
#pragma unroll
    for (int i = 0; i < (BM * BK) / 256; ++i) {   // 16 halves/thread
      int idx = tid + i * 256;
      int r = idx >> 5, c = idx & 31;
      size_t g = (size_t)(blockM + r) * K + (k0 + c);
      Al[r][c] = Alo[g];
      if (SPLIT) Ah[r][c] = Ahi[g];
    }
#pragma unroll
    for (int i = 0; i < (BK * BN) / 256; ++i) {   // 8 halves/thread
      int idx = tid + i * 256;
      int r = idx >> 6, c = idx & 63;
      Bsh[r][c] = B[(size_t)(k0 + r) * Nn + (blockN + c)];
    }
    __syncthreads();

    // ---- fragment gather per ISA 16-bit lane maps ----
    v16h afL[2], afH[2], bf[2];
#pragma unroll
    for (int tm = 0; tm < 2; ++tm) {
      int row = wm * 32 + tm * 16 + l16;
#pragma unroll
      for (int v = 0; v < 8; ++v) {
        // A 16x32 map: VGPR v<4 -> K=8*half+2v ; v>=4 -> K=16+8*half+2(v-4)
        int kb = (v < 4) ? (8 * half + 2 * v) : (16 + 8 * half + 2 * (v - 4));
        afL[tm][2 * v + 0] = Al[row][kb + 0];
        afL[tm][2 * v + 1] = Al[row][kb + 1];
        if (SPLIT) {
          afH[tm][2 * v + 0] = Ah[row][kb + 0];
          afH[tm][2 * v + 1] = Ah[row][kb + 1];
        }
      }
    }
#pragma unroll
    for (int tn = 0; tn < 2; ++tn) {
      int col = wn * 32 + tn * 16 + l16;
#pragma unroll
      for (int j = 0; j < 16; ++j)   // B 32x16 map: j<8 -> K=8*half+j ; j>=8 -> K=16+8*half+(j-8)
        bf[tn][j] = Bsh[(j < 8) ? (8 * half + j) : (16 + 8 * half + (j - 8))][col];
    }

#pragma unroll
    for (int tm = 0; tm < 2; ++tm)
#pragma unroll
      for (int tn = 0; tn < 2; ++tn) {
        accL[tm][tn] = __builtin_amdgcn_wmma_f32_16x16x32_f16(
            false, afL[tm], false, bf[tn], (short)0, accL[tm][tn], false, false);
        if (SPLIT)
          accH[tm][tn] = __builtin_amdgcn_wmma_f32_16x16x32_f16(
              false, afH[tm], false, bf[tn], (short)0, accH[tm][tn], false, false);
      }
    __syncthreads();
  }

#pragma unroll
  for (int tm = 0; tm < 2; ++tm)
#pragma unroll
    for (int tn = 0; tn < 2; ++tn) {
      int col = blockN + wn * 32 + tn * 16 + l16;
#pragma unroll
      for (int g = 0; g < 8; ++g) {
        int row = blockM + wm * 32 + tm * 16 + g + 8 * half;
        float v = SPLIT ? 2048.0f * accH[tm][tn][g] + accL[tm][tn][g]
                        : accL[tm][tn][g];
        C[(size_t)row * Nn + col] = v;
      }
    }
}

// ---------------------------------------------------------------------------
// Elementwise helpers
// ---------------------------------------------------------------------------
__global__ void fill_zero_k(float* __restrict__ p, size_t n) {
  size_t i = (size_t)blockIdx.x * blockDim.x + threadIdx.x;
  size_t stride = (size_t)gridDim.x * blockDim.x;
  for (; i < n; i += stride) p[i] = 0.0f;
}

__global__ void scatter_edges_k(const int* __restrict__ ei, float* __restrict__ adj,
                                int E, int n) {
  int e = blockIdx.x * blockDim.x + threadIdx.x;
  if (e < E) {
    int src = ei[e];
    int dst = ei[E + e];
    adj[(size_t)dst * n + src] = 1.0f;
  }
}

__global__ void cvt_f16_k(_Float16* __restrict__ dst, const float* __restrict__ src,
                          size_t n) {
  size_t i = (size_t)blockIdx.x * blockDim.x + threadIdx.x;
  size_t stride = (size_t)gridDim.x * blockDim.x;
  for (; i < n; i += stride) dst[i] = (_Float16)src[i];
}

// v (non-negative integer < 2^22) -> hi = floor(v/2048), lo = v - 2048*hi (exact)
__global__ void split_hi_lo_k(const float* __restrict__ P, _Float16* __restrict__ hi,
                              _Float16* __restrict__ lo, size_t n) {
  size_t i = (size_t)blockIdx.x * blockDim.x + threadIdx.x;
  size_t stride = (size_t)gridDim.x * blockDim.x;
  for (; i < n; i += stride) {
    float v = P[i];
    float h = floorf(v * (1.0f / 2048.0f));
    hi[i] = (_Float16)h;
    lo[i] = (_Float16)(v - 2048.0f * h);
  }
}

__global__ void coeffs_k(const float* __restrict__ w1, const float* __restrict__ w2,
                         float* __restrict__ c) {
  if (threadIdx.x == 0 && blockIdx.x == 0) {
    float p1 = 1.0f, p2 = 1.0f;
    for (int i = 0; i <= LPOW; ++i) {
      p1 *= w1[i]; c[i] = p1;
      p2 *= w2[i]; c[8 + i] = p2;
    }
  }
}

__global__ void init_S_k(float* __restrict__ S, const float* __restrict__ adj,
                         const float* __restrict__ c, int n) {
  size_t total = (size_t)n * n;
  size_t i = (size_t)blockIdx.x * blockDim.x + threadIdx.x;
  size_t stride = (size_t)gridDim.x * blockDim.x;
  float c0 = c[0], c1 = c[1];
  for (; i < total; i += stride) {
    int r = (int)(i / n), col = (int)(i % n);
    S[i] = c1 * adj[i] + (r == col ? c0 : 0.0f);
  }
}

__global__ void accum_S_k(float* __restrict__ S, const float* __restrict__ P,
                          const float* __restrict__ c, int ci, size_t total) {
  size_t i = (size_t)blockIdx.x * blockDim.x + threadIdx.x;
  size_t stride = (size_t)gridDim.x * blockDim.x;
  float cv = c[ci];
  for (; i < total; i += stride) S[i] += cv * P[i];
}

__global__ __launch_bounds__(256) void row_dis_k(const float* __restrict__ S,
                                                 float* __restrict__ dis, int n) {
  __shared__ int red[256];
  int r = blockIdx.x;
  int cnt = 0;
  for (int c = threadIdx.x; c < n; c += 256)
    cnt += (S[(size_t)r * n + c] > 0.0f) ? 1 : 0;
  red[threadIdx.x] = cnt;
  __syncthreads();
  for (int s = 128; s > 0; s >>= 1) {
    if (threadIdx.x < s) red[threadIdx.x] += red[threadIdx.x + s];
    __syncthreads();
  }
  if (threadIdx.x == 0) {
    float d = (float)red[0];
    dis[r] = (d > 0.0f) ? 1.0f / sqrtf(d) : 0.0f;
  }
}

__global__ void scale_rows_k(float* __restrict__ dst, const float* __restrict__ src,
                             const float* __restrict__ s, int rows, int cols) {
  size_t total = (size_t)rows * cols;
  size_t i = (size_t)blockIdx.x * blockDim.x + threadIdx.x;
  size_t stride = (size_t)gridDim.x * blockDim.x;
  for (; i < total; i += stride) dst[i] = src[i] * s[i / cols];
}

__global__ void bias_relu_k(float* __restrict__ Y, const float* __restrict__ X,
                            const float* __restrict__ b, const float* __restrict__ rscale,
                            int rows, int cols) {
  size_t total = (size_t)rows * cols;
  size_t i = (size_t)blockIdx.x * blockDim.x + threadIdx.x;
  size_t stride = (size_t)gridDim.x * blockDim.x;
  for (; i < total; i += stride) {
    float v = X[i];
    if (rscale) v *= rscale[i / cols];
    v += b[i % cols];
    Y[i] = v > 0.0f ? v : 0.0f;
  }
}

// ---------------------------------------------------------------------------
static inline void launch_gemm_nn(const float* A, const float* B, float* C,
                                  int M, int Nn, int K, hipStream_t s) {
  dim3 grid(Nn / 64, M / 128);
  gemm_f32_wmma<false><<<grid, 256, 0, s>>>(A, B, C, M, Nn, K);
}
static inline void launch_gemm_nt(const float* A, const float* B, float* C,
                                  int M, int Nn, int K, hipStream_t s) {
  dim3 grid(Nn / 64, M / 128);
  gemm_f32_wmma<true><<<grid, 256, 0, s>>>(A, B, C, M, Nn, K);
}

extern "C" void kernel_launch(void* const* d_in, const int* in_sizes, int n_in,
                              void* d_out, int out_size, void* d_ws, size_t ws_size,
                              hipStream_t stream) {
  const int n   = NNODES;
  const size_t NN = (size_t)n * n;
  const int E   = in_sizes[1] / 2;

  const float* x       = (const float*)d_in[0];
  const int*   ei      = (const int*)  d_in[1];
  const float* w1_path = (const float*)d_in[2];
  const float* lin1_w  = (const float*)d_in[3];
  const float* lin1_b  = (const float*)d_in[4];
  const float* w2_path = (const float*)d_in[5];
  const float* lin2_w  = (const float*)d_in[6];
  const float* lin2_b  = (const float*)d_in[7];
  float* out = (float*)d_out;

  // -------- workspace layout (floats); H overlays adj/P0/P1 region --------
  float* ws = (float*)d_ws;
  float* S1   = ws;                          // NN
  float* S2   = S1 + NN;                     // NN
  float* APP  = S2 + NN;                     // max(3*NN, N*HID)
  float* adj  = APP;                         // NN  (phase A)
  float* P0   = APP + NN;                    // NN  (phase A)
  float* P1   = APP + 2 * NN;                // NN  (phase A)
  float* H    = APP;                         // N*HID (phase B overlay)
  float* smallbuf = APP + (size_t)n * HIDC;
  float* coef = smallbuf;                    // 16
  float* dis1 = coef + 16;                   // n
  float* dis2 = dis1 + n;                    // n
  float* xs   = dis2 + n;                    // n*INC
  float* T1   = xs + (size_t)n * INC;        // n*INC
  float* G    = T1 + (size_t)n * INC;        // n*OUTC
  float* T2   = G  + (size_t)n * OUTC;       // n*OUTC
  // f16 buffers for the split power chain (after the float region)
  _Float16* adjf = (_Float16*)(T2 + (size_t)n * OUTC);   // NN halves
  _Float16* Ahi  = adjf + NN;                            // NN halves
  _Float16* Alo  = Ahi + NN;                             // NN halves

  const int TB = 256;
  const int GB_NN = (int)((NN + TB - 1) / TB);
  dim3 pow_grid(n / 64, n / 128);

  // -------- phase A: adjacency, powers (split-f16 WMMA), S1/S2 --------
  fill_zero_k<<<GB_NN, TB, 0, stream>>>(adj, NN);
  scatter_edges_k<<<(E + TB - 1) / TB, TB, 0, stream>>>(ei, adj, E, n);
  coeffs_k<<<1, 64, 0, stream>>>(w1_path, w2_path, coef);
  cvt_f16_k<<<GB_NN, TB, 0, stream>>>(adjf, adj, NN);

  init_S_k<<<GB_NN, TB, 0, stream>>>(S1, adj, coef + 0, n);
  init_S_k<<<GB_NN, TB, 0, stream>>>(S2, adj, coef + 8, n);

  // A^2 = adjf @ adjf (binary x binary, no split needed)
  gemm_pow_f16_wmma<false><<<pow_grid, 256, 0, stream>>>(adjf, adjf, adjf, P0, n, n, n);
  accum_S_k<<<GB_NN, TB, 0, stream>>>(S1, P0, coef, 2, NN);
  accum_S_k<<<GB_NN, TB, 0, stream>>>(S2, P0, coef, 10, NN);

  // A^3..A^5: exact split-f16 (A = 2048*hi + lo, both f16-exact integers)
  split_hi_lo_k<<<GB_NN, TB, 0, stream>>>(P0, Ahi, Alo, NN);
  gemm_pow_f16_wmma<true><<<pow_grid, 256, 0, stream>>>(Ahi, Alo, adjf, P1, n, n, n);
  accum_S_k<<<GB_NN, TB, 0, stream>>>(S1, P1, coef, 3, NN);
  accum_S_k<<<GB_NN, TB, 0, stream>>>(S2, P1, coef, 11, NN);

  split_hi_lo_k<<<GB_NN, TB, 0, stream>>>(P1, Ahi, Alo, NN);
  gemm_pow_f16_wmma<true><<<pow_grid, 256, 0, stream>>>(Ahi, Alo, adjf, P0, n, n, n);
  accum_S_k<<<GB_NN, TB, 0, stream>>>(S1, P0, coef, 4, NN);
  accum_S_k<<<GB_NN, TB, 0, stream>>>(S2, P0, coef, 12, NN);

  split_hi_lo_k<<<GB_NN, TB, 0, stream>>>(P0, Ahi, Alo, NN);
  gemm_pow_f16_wmma<true><<<pow_grid, 256, 0, stream>>>(Ahi, Alo, adjf, P1, n, n, n);
  accum_S_k<<<GB_NN, TB, 0, stream>>>(S1, P1, coef, 5, NN);
  accum_S_k<<<GB_NN, TB, 0, stream>>>(S2, P1, coef, 13, NN);

  // degree -> dis
  row_dis_k<<<n, 256, 0, stream>>>(S1, dis1, n);
  row_dis_k<<<n, 256, 0, stream>>>(S2, dis2, n);

  // -------- phase B: layer 1 (fp32 WMMA, exact value path) --------
  scale_rows_k<<<(int)(((size_t)n * INC + TB - 1) / TB), TB, 0, stream>>>(xs, x, dis1, n, INC);
  launch_gemm_nn(S1, xs, T1, n, INC, n, stream);
  scale_rows_k<<<(int)(((size_t)n * INC + TB - 1) / TB), TB, 0, stream>>>(T1, T1, dis1, n, INC);
  launch_gemm_nt(T1, lin1_w, H, n, HIDC, INC, stream);
  bias_relu_k<<<(int)(((size_t)n * HIDC + TB - 1) / TB), TB, 0, stream>>>(
      H, H, lin1_b, nullptr, n, HIDC);

  // -------- layer 2 (reassociated: M2 @ (H @ W2^T) + b2) --------
  launch_gemm_nt(H, lin2_w, G, n, OUTC, HIDC, stream);
  scale_rows_k<<<(int)(((size_t)n * OUTC + TB - 1) / TB), TB, 0, stream>>>(G, G, dis2, n, OUTC);
  launch_gemm_nn(S2, G, T2, n, OUTC, n, stream);
  bias_relu_k<<<(int)(((size_t)n * OUTC + TB - 1) / TB), TB, 0, stream>>>(
      out, T2, lin2_b, dis2, n, OUTC);

  (void)n_in; (void)out_size; (void)ws_size;
}
